// GemmaAttention_74895639708232
// MI455X (gfx1250) — matrix operand order repaired and checked by reading, not compiled, not run on hardware
//
#include <hip/hip_runtime.h>
#include <hip/hip_bf16.h>

// ---------------------------------------------------------------------------
// Types
// ---------------------------------------------------------------------------
typedef __bf16 bf16_t;
typedef __bf16 bf16x4  __attribute__((ext_vector_type(4)));
typedef __bf16 bf16x8  __attribute__((ext_vector_type(8)));
typedef __bf16 bf16x16 __attribute__((ext_vector_type(16)));
typedef float  f32x4   __attribute__((ext_vector_type(4)));
typedef float  f32x8   __attribute__((ext_vector_type(8)));
typedef unsigned int u32x4 __attribute__((ext_vector_type(4)));
typedef int    i32x4   __attribute__((ext_vector_type(4)));
typedef int    i32x8   __attribute__((ext_vector_type(8)));

#define B_     4
#define S_     2048
#define H_     2048
#define NH_    8
#define D_     256
#define QKV_N  2560            // (NH + 2*NKV) * D
#define SCALE_ 0.0625f         // 256^-0.5
#define MASKV  (-3.0e38f)

union FragAB { bf16x16 v; bf16x8 h[2]; };

// ---------------------------------------------------------------------------
// Tensor Data Mover: 2D bf16 tile Global -> LDS, with LDS row padding.
// D# built per CDNA5 ISA 8.3/8.4. One instruction per wave; TENSORcnt-tracked.
//   pad_interval_code: trigger every 2^(code+1) DWORDs; pad_amount_code: +code+1 DWORDs
// ---------------------------------------------------------------------------
__device__ __forceinline__ void tdm_load_2d_bf16(
    unsigned lds_off, const void* gptr,
    unsigned tensor_d0, unsigned tensor_d1,
    unsigned tile_d0, unsigned tile_d1, unsigned stride0,
    unsigned pad_interval_code, unsigned pad_amount_code)
{
  unsigned long long ga = (unsigned long long)(uintptr_t)gptr;
  u32x4 g0;
  g0.x = 1u;                                           // count=1, user mode
  g0.y = lds_off;                                      // lds_addr (bytes)
  g0.z = (unsigned)ga;                                 // global_addr[31:0]
  g0.w = (unsigned)((ga >> 32) & 0x01FFFFFFu) | (2u << 30);  // addr[56:32] | type=2
  i32x8 g1;
  g1[0] = (int)((1u << 16)                             // data_size=1 -> 2 bytes
              | (1u << 20)                             // pad_enable
              | (pad_interval_code << 22)
              | (pad_amount_code << 25));
  g1[1] = (int)((tensor_d0 & 0xFFFFu) << 16);          // tensor_dim0[15:0] @ bits63:48
  g1[2] = (int)((tensor_d0 >> 16) | ((tensor_d1 & 0xFFFFu) << 16));
  g1[3] = (int)((tensor_d1 >> 16) | (tile_d0 << 16));  // tile_dim0 @ bits127:112
  g1[4] = (int)tile_d1;                                // tile_dim1; tile_dim2=0
  g1[5] = (int)stride0;                                // tensor_dim0_stride[31:0]
  g1[6] = 0;
  g1[7] = 0;
  i32x4 z4;
  z4[0] = 0; z4[1] = 0; z4[2] = 0; z4[3] = 0;
  i32x8 z8;
  for (int i = 0; i < 8; i++) z8[i] = 0;
  // amdgpu-toolchain (clang-23) 6-arg form: (g0, g1, g2, g3, extra, cpol)
  __builtin_amdgcn_tensor_load_to_lds(g0, g1, z4, z4, z8, 0);
}

// ---------------------------------------------------------------------------
// GEMM: C[M,N] = A[M,K] * B[K,N], bf16 WMMA, f32 accumulate.
// Block: 256 threads (8 waves), tile 128(M) x 64(N), BK=32.
// A_BF16 path stages A via the Tensor Data Mover (pad -> 96B LDS rows).
// ---------------------------------------------------------------------------
template <bool A_BF16, bool STORE_BF16>
__global__ __launch_bounds__(256) void gemm_wmma(
    const void* __restrict__ Araw, const float* __restrict__ Bmat,
    void* __restrict__ Cmat, int M, int N, int K)
{
  __shared__ bf16_t As[128 * 48];   // row-major, stride 48 (96B rows)
  __shared__ bf16_t Bs[64 * 48];    // transposed: Bs[n][k], stride 48

  const int tid  = threadIdx.x;
  const int wave = tid >> 5;
  const int lane = tid & 31;
  const int m    = lane & 15;
  const int hi   = lane >> 4;
  const int m0   = blockIdx.y * 128;
  const int n0   = blockIdx.x * 64;

  f32x8 acc[4];
  for (int nt = 0; nt < 4; nt++)
    for (int i = 0; i < 8; i++) acc[nt][i] = 0.0f;

  for (int kk = 0; kk < K; kk += 32) {
    // ---- stage A tile (128 x 32) ----
    if constexpr (A_BF16) {
      // TDM: 128 rows x 64B, +32B LDS pad per row -> stride 96B (= As stride 48)
      if (wave == 0) {
        tdm_load_2d_bf16((unsigned)(uintptr_t)(void*)As,
                         (const bf16_t*)Araw + (size_t)m0 * K + kk,
                         /*tensor_d0=*/32, /*tensor_d1=*/128,
                         /*tile_d0=*/32, /*tile_d1=*/128,
                         /*stride0=*/(unsigned)K,
                         /*pad: every 16 DW (64B)*/ 3, /*+8 DW (32B)*/ 7);
      }
    } else {
      const float* A = (const float*)Araw;
      #pragma unroll
      for (int f = tid; f < 1024; f += 256) {            // 1024 float4 units
        int r = f >> 3, c4 = (f & 7) * 4;
        f32x4 v = *(const f32x4*)(A + (size_t)(m0 + r) * K + kk + c4);
        bf16x4 bv;
        for (int i = 0; i < 4; i++) bv[i] = (bf16_t)v[i];
        *(bf16x4*)(As + r * 48 + c4) = bv;
      }
    }
    // ---- stage B tile (32 x 64), transposed into Bs[n][k] ----
    #pragma unroll
    for (int f = tid; f < 512; f += 256) {               // 512 float4 units
      int r = f >> 4, c4 = (f & 15) * 4;
      f32x4 v = *(const f32x4*)(Bmat + (size_t)(kk + r) * N + n0 + c4);
      for (int i = 0; i < 4; i++) Bs[(c4 + i) * 48 + r] = (bf16_t)v[i];
    }
    // prefetch next B tile into cache (global_prefetch_b8)
    if (kk + 32 < K)
      __builtin_prefetch(Bmat + (size_t)(kk + 32 + (tid >> 3)) * N + n0 + (tid & 7) * 8, 0, 1);

    if constexpr (A_BF16) {
      if (wave == 0) __builtin_amdgcn_s_wait_tensorcnt(0);
    }
    __syncthreads();

    // ---- fragments + WMMA ----
    FragAB af;
    const bf16_t* arow = As + (wave * 16 + m) * 48 + hi * 8;
    af.h[0] = *(const bf16x8*)(arow);
    af.h[1] = *(const bf16x8*)(arow + 16);
    #pragma unroll
    for (int nt = 0; nt < 4; nt++) {
      FragAB bfr;
      const bf16_t* brow = Bs + (nt * 16 + m) * 48 + hi * 8;
      bfr.h[0] = *(const bf16x8*)(brow);
      bfr.h[1] = *(const bf16x8*)(brow + 16);
      acc[nt] = __builtin_amdgcn_wmma_f32_16x16x32_bf16(
          false, af.v, false, bfr.v, (short)0, acc[nt], false, false);
    }
    __syncthreads();
  }

  // ---- epilogue: C layout = lane -> col, VGPR r -> row (hi*8 + r) ----
  const int row = m0 + wave * 16 + hi * 8;
  #pragma unroll
  for (int nt = 0; nt < 4; nt++) {
    const int col = n0 + nt * 16 + m;
    if (STORE_BF16) {
      bf16_t* C = (bf16_t*)Cmat;
      for (int r = 0; r < 8; r++)
        C[(size_t)(row + r) * N + col] = (bf16_t)acc[nt][r];
    } else {
      float* C = (float*)Cmat;
      for (int r = 0; r < 8; r++)
        C[(size_t)(row + r) * N + col] = acc[nt][r];
    }
  }
}

// ---------------------------------------------------------------------------
// RoPE + split qkv -> q (B,NH,S,D), k (B,S,D), v (B,S,D), all bf16.
// ---------------------------------------------------------------------------
__global__ __launch_bounds__(128) void rope_split(
    const bf16_t* __restrict__ qkv, const int* __restrict__ positions,
    bf16_t* __restrict__ qout, bf16_t* __restrict__ kout,
    bf16_t* __restrict__ vout)
{
  const int row = blockIdx.x;            // b*S + s
  const int j   = threadIdx.x;           // 0..127
  const int s   = row & (S_ - 1);
  const int b   = row >> 11;

  const float pos = (float)positions[s];
  const float inv = __expf(-(float)(2 * j) * (9.210340371976184f / 256.0f));
  const float ang = pos * inv;
  float sn, cs;
  __sincosf(ang, &sn, &cs);

  const bf16_t* qr = qkv + (size_t)row * QKV_N;
  #pragma unroll
  for (int h = 0; h < NH_; h++) {
    float x1 = (float)qr[h * 256 + j];
    float x2 = (float)qr[h * 256 + 128 + j];
    bf16_t* dst = qout + (((size_t)(b * NH_ + h)) * S_ + s) * D_;
    dst[j]       = (bf16_t)(x1 * cs - x2 * sn);
    dst[j + 128] = (bf16_t)(x2 * cs + x1 * sn);
  }
  {
    float x1 = (float)qr[2048 + j];
    float x2 = (float)qr[2048 + 128 + j];
    bf16_t* dst = kout + (size_t)row * D_;
    dst[j]       = (bf16_t)(x1 * cs - x2 * sn);
    dst[j + 128] = (bf16_t)(x2 * cs + x1 * sn);
  }
  vout[(size_t)row * D_ + j]       = qr[2304 + j];
  vout[(size_t)row * D_ + 128 + j] = qr[2304 + 128 + j];
}

// ---------------------------------------------------------------------------
// Flash attention with causal mask, bf16 WMMA, online softmax.
// Grid: (S/128, B*NH). Block: 256 threads = 8 waves; wave w owns 16 q-rows.
// K block staged by the Tensor Data Mover (pad -> 528B LDS rows); V staged
// transposed manually; Q fragments re-read from global (L2-resident).
// ---------------------------------------------------------------------------
__global__ __launch_bounds__(256) void flash_attn(
    const bf16_t* __restrict__ Q, const bf16_t* __restrict__ Kc,
    const bf16_t* __restrict__ Vc, bf16_t* __restrict__ Oattn)
{
  __shared__ bf16_t Ks[32 * 264];        // K rows, stride 264 (528B rows)
  __shared__ bf16_t Vt[256 * 40];        // V transposed: Vt[d][kpos], stride 40
  __shared__ bf16_t Ps[8 * 16 * 32];     // per-wave P staging (bf16)

  const int tid  = threadIdx.x;
  const int wave = tid >> 5;
  const int lane = tid & 31;
  const int m    = lane & 15;
  const int hi   = lane >> 4;
  const int bh   = blockIdx.y;
  const int b    = bh >> 3;
  const int h    = bh & 7;
  const int q0   = blockIdx.x * 128;

  const bf16_t* qrow = Q + (((size_t)bh) * S_ + q0 + wave * 16 + m) * D_;

  f32x8 o[16];
  for (int nt = 0; nt < 16; nt++)
    for (int i = 0; i < 8; i++) o[nt][i] = 0.0f;
  float mrow[8], lrow[8];
  for (int r = 0; r < 8; r++) { mrow[r] = MASKV; lrow[r] = 0.0f; }

  const bf16_t* kbase = Kc + (size_t)b * S_ * D_;
  const bf16_t* vbase = Vc + (size_t)b * S_ * D_;
  const int rowg  = q0 + wave * 16 + hi * 8;   // +r gives global q row
  const int nblk  = (q0 + 128) / 32;

  for (int blk = 0; blk < nblk; blk++) {
    const int kb = blk * 32;
    // ---- TDM: stage K rows kb..kb+31 (512B rows + 16B pad -> stride 264) ----
    if (wave == 0) {
      tdm_load_2d_bf16((unsigned)(uintptr_t)(void*)Ks,
                       kbase + (size_t)kb * D_,
                       /*tensor_d0=*/256, /*tensor_d1=*/32,
                       /*tile_d0=*/256, /*tile_d1=*/32,
                       /*stride0=*/256,
                       /*pad: every 128 DW (512B)*/ 6, /*+4 DW (16B)*/ 3);
    }
    // ---- stage V rows transposed: Vt[d][kpos] ----
    #pragma unroll
    for (int f = tid; f < 1024; f += 256) {
      int r = f >> 5, c8 = (f & 31) * 8;
      bf16x8 v = *(const bf16x8*)(vbase + (size_t)(kb + r) * D_ + c8);
      for (int i = 0; i < 8; i++) Vt[(c8 + i) * 40 + r] = v[i];
    }
    if (wave == 0) __builtin_amdgcn_s_wait_tensorcnt(0);
    __syncthreads();

    // ---- scores S = Q * K^T  (16 x 32 per wave, two 16x16 tiles) ----
    f32x8 st[2];
    #pragma unroll
    for (int j = 0; j < 2; j++) {
      f32x8 a;
      for (int i = 0; i < 8; i++) a[i] = 0.0f;
      const bf16_t* krow = Ks + (j * 16 + m) * 264 + hi * 8;
      #pragma unroll
      for (int c = 0; c < 8; c++) {
        FragAB qf, kf;
        const int base = c * 32 + hi * 8;
        qf.h[0] = *(const bf16x8*)(qrow + base);
        qf.h[1] = *(const bf16x8*)(qrow + base + 16);
        kf.h[0] = *(const bf16x8*)(krow + c * 32);
        kf.h[1] = *(const bf16x8*)(krow + c * 32 + 16);
        a = __builtin_amdgcn_wmma_f32_16x16x32_bf16(
            false, qf.v, false, kf.v, (short)0, a, false, false);
      }
      st[j] = a;
    }

    // ---- scale + causal mask (C layout: lane->kpos, VGPR r->row) ----
    #pragma unroll
    for (int j = 0; j < 2; j++) {
      const int kpos = kb + j * 16 + m;
      for (int r = 0; r < 8; r++) {
        float sc = st[j][r] * SCALE_;
        st[j][r] = (kpos <= rowg + r) ? sc : MASKV;
      }
    }

    // ---- online softmax: row max / rescale / row sum ----
    float mnew[8], pscale[8];
    #pragma unroll
    for (int r = 0; r < 8; r++) {
      float v = fmaxf(st[0][r], st[1][r]);
      for (int msk = 1; msk < 16; msk <<= 1)
        v = fmaxf(v, __shfl_xor(v, msk, 32));
      mnew[r]   = fmaxf(mrow[r], v);
      pscale[r] = __expf(mrow[r] - mnew[r]);
      mrow[r]   = mnew[r];
    }
    #pragma unroll
    for (int j = 0; j < 2; j++)
      for (int r = 0; r < 8; r++)
        st[j][r] = __expf(st[j][r] - mnew[r]);
    #pragma unroll
    for (int r = 0; r < 8; r++) {
      float v = st[0][r] + st[1][r];
      for (int msk = 1; msk < 16; msk <<= 1)
        v += __shfl_xor(v, msk, 32);
      lrow[r] = lrow[r] * pscale[r] + v;
    }

    // ---- restage P as bf16 through per-wave LDS (C layout -> A layout) ----
    bf16_t* pw = Ps + wave * (16 * 32);
    #pragma unroll
    for (int j = 0; j < 2; j++)
      for (int r = 0; r < 8; r++)
        pw[(hi * 8 + r) * 32 + j * 16 + m] = (bf16_t)st[j][r];

    // ---- rescale O while the LDS writes drain ----
    #pragma unroll
    for (int nt = 0; nt < 16; nt++)
      for (int r = 0; r < 8; r++) o[nt][r] *= pscale[r];

    // wave-local RAW on our own Ps region: order DS ops, no wg barrier needed
    asm volatile("s_wait_dscnt 0x0" ::: "memory");

    // ---- O += P * V ----
    FragAB pf;
    const bf16_t* prow = pw + m * 32 + hi * 8;
    pf.h[0] = *(const bf16x8*)(prow);
    pf.h[1] = *(const bf16x8*)(prow + 16);
    #pragma unroll
    for (int nt = 0; nt < 16; nt++) {
      FragAB vf;
      const bf16_t* vrow = Vt + (nt * 16 + m) * 40 + hi * 8;
      vf.h[0] = *(const bf16x8*)(vrow);
      vf.h[1] = *(const bf16x8*)(vrow + 16);
      o[nt] = __builtin_amdgcn_wmma_f32_16x16x32_bf16(
          false, pf.v, false, vf.v, (short)0, o[nt], false, false);
    }
    __syncthreads();   // everyone done reading Ks/Vt before restage
  }

  // ---- normalize + store attn (B, S, NH*D) as bf16 ----
  float inv_l[8];
  for (int r = 0; r < 8; r++) inv_l[r] = 1.0f / lrow[r];
  #pragma unroll
  for (int nt = 0; nt < 16; nt++) {
    const int col = h * 256 + nt * 16 + m;
    for (int r = 0; r < 8; r++)
      Oattn[((size_t)(b * S_ + rowg + r)) * (NH_ * D_) + col] =
          (bf16_t)(o[nt][r] * inv_l[r]);
  }
}

// ---------------------------------------------------------------------------
// Launch
// ---------------------------------------------------------------------------
extern "C" void kernel_launch(void* const* d_in, const int* in_sizes, int n_in,
                              void* d_out, int out_size, void* d_ws, size_t ws_size,
                              hipStream_t stream) {
  const float* hs   = (const float*)d_in[0];
  const int*   pos  = (const int*)d_in[1];
  const float* wqkv = (const float*)d_in[2];
  const float* wo   = (const float*)d_in[3];
  float*       out  = (float*)d_out;

  char* ws = (char*)d_ws;
  const size_t qkv_bytes = (size_t)B_ * S_ * QKV_N * 2;       // 41,943,040
  const size_t q_bytes   = (size_t)B_ * NH_ * S_ * D_ * 2;    // 33,554,432
  const size_t kv_bytes  = (size_t)B_ * S_ * D_ * 2;          //  4,194,304

  bf16_t* qkv_bf  = (bf16_t*)ws;                 // consumed by rope
  bf16_t* attn_bf = (bf16_t*)ws;                 // aliases qkv region (safe)
  bf16_t* q_bf    = (bf16_t*)(ws + qkv_bytes);
  bf16_t* k_bf    = (bf16_t*)(ws + qkv_bytes + q_bytes);
  bf16_t* v_bf    = (bf16_t*)(ws + qkv_bytes + q_bytes + kv_bytes);

  // 1) QKV projection: (8192 x 2048) * (2048 x 2560) -> bf16 qkv
  gemm_wmma<false, true><<<dim3(QKV_N / 64, (B_ * S_) / 128), dim3(256), 0, stream>>>(
      (const void*)hs, wqkv, (void*)qkv_bf, B_ * S_, QKV_N, H_);

  // 2) RoPE + split into q/k/v (bf16)
  rope_split<<<dim3(B_ * S_), dim3(128), 0, stream>>>(qkv_bf, pos, q_bf, k_bf, v_bf);

  // 3) Causal flash attention -> attn bf16 (B, S, NH*D)
  flash_attn<<<dim3(S_ / 128, B_ * NH_), dim3(256), 0, stream>>>(
      q_bf, k_bf, v_bf, attn_bf);

  // 4) Output projection: (8192 x 2048) * (2048 x 2048) -> fp32 out
  gemm_wmma<true, false><<<dim3(H_ / 64, (B_ * S_) / 128), dim3(256), 0, stream>>>(
      (const void*)attn_bf, wo, (void*)out, B_ * S_, H_, NH_ * D_);
}
